// Windowed3DAttn_32177894981705
// MI455X (gfx1250) — matrix operand order repaired and checked by reading, not compiled, not run on hardware
//
#include <hip/hip_runtime.h>
#include <stdint.h>

// Problem constants (B=1)
#define D_   32
#define H_   64
#define W_   128
#define C_   96
#define SLEN 64      // 4*4*4 tokens per window
#define CH   24      // channels per head
#define NWIN 4096    // 8*16*32 windows

typedef __attribute__((ext_vector_type(16))) __bf16 v16bf;
typedef __attribute__((ext_vector_type(8)))  float  v8f;

union Frag16 { v16bf v; unsigned u[8]; unsigned short h[16]; };

// float -> bf16 (RNE) through the native type so the backend can use
// hardware cvt instructions instead of a 4-op integer rounding sequence.
__device__ __forceinline__ unsigned short f2bf(float f) {
  __bf16 h = (__bf16)f;
  return __builtin_bit_cast(unsigned short, h);
}

__device__ __forceinline__ v8f wmma_bf16(v16bf a, v16bf b, v8f c) {
  return __builtin_amdgcn_wmma_f32_16x16x32_bf16(false, a, false, b, (short)0, c, false, false);
}

// A-matrix (16x32 bf16) gather from row-major LDS [rows][stride] (elements).
// lane L(0..15): M=row0+L ; VGPR j: K = (L>=16 ? 16:0) + 2*j, +1
__device__ __forceinline__ v16bf ldsA(const unsigned short* p0, int stride,
                                      int row0, int col0, int kmax, int lane) {
  Frag16 f;
  const int hf = lane >> 4;
  const unsigned short* p = p0 + (row0 + (lane & 15)) * stride + col0;
#pragma unroll
  for (int j = 0; j < 8; ++j) {
    int kk = hf * 16 + 2 * j;
    f.u[j] = (kk < kmax) ? *(const unsigned*)(p + kk) : 0u;
  }
  return f.v;
}

// B-matrix (32x16 bf16) from K-contiguous ("transposed") LDS: elem(k,n)=p0[n*stride+col0+k]
// lane L(0..15): N=n0+L, VGPR j: K=2j,2j+1 ; lanes 16..31: K += 16
__device__ __forceinline__ v16bf ldsBT(const unsigned short* p0, int stride,
                                       int n0, int col0, int kmax, int lane) {
  Frag16 f;
  const int hf = lane >> 4;
  const unsigned short* p = p0 + (n0 + (lane & 15)) * stride + col0;
#pragma unroll
  for (int j = 0; j < 8; ++j) {
    int kk = hf * 16 + 2 * j;
    f.u[j] = (kk < kmax) ? *(const unsigned*)(p + kk) : 0u;
  }
  return f.v;
}

// B-matrix (32x16 bf16) from row-major-in-K LDS: elem(k,n)=p0[(k0+k)*stride+col]
__device__ __forceinline__ v16bf ldsBrm(const unsigned short* p0, int stride,
                                        int k0, int col, int valid, int lane) {
  Frag16 f;
  const int hf = lane >> 4;
  const unsigned short* p = p0 + (k0 + hf * 16) * stride + col;
#pragma unroll
  for (int j = 0; j < 8; ++j) {
    unsigned lo = valid ? (unsigned)p[(2 * j) * stride]     : 0u;
    unsigned hi = valid ? (unsigned)p[(2 * j + 1) * stride] : 0u;
    f.u[j] = lo | (hi << 16);
  }
  return f.v;
}

// window id + token -> flat index into [D,H,W,C]
__device__ __forceinline__ int gidx(int wid, int t, int c) {
  int iz = wid >> 9;          // / (16*32)
  int iy = (wid >> 5) & 15;
  int ix = wid & 31;
  int z = iz * 4 + (t >> 4);
  int y = iy * 4 + ((t >> 2) & 3);
  int xx = ix * 4 + (t & 3);
  return ((z * H_ + y) * W_ + xx) * C_ + c;
}

// ---- LDS layout (bytes) ----
#define QS   296                      // QKV row stride (elements)
#define XS   104                      // X / O row stride
#define PS   72                       // P row stride
#define OFF_WQT 0                     // [288][96] bf16 = 55296 B (WqkvT)
#define OFF_WPT 55296                 // [96][96]  bf16 = 18432 B (WprojT)
#define OFF_QKV 73728                 // [64][296] bf16 = 37888 B
#define OFF_R1  111616                // X [64][104] (13312 B) then P 4*[64][72] (36864 B)
#define OFF_O   148480                // O [64][104] bf16 = 13312 B
#define SMEM_BYTES 161792

// Prep: transpose weights to [n][k] bf16 in workspace (wqkvT then wprojT, contiguous)
__global__ __launch_bounds__(256) void prep_weights(const float* __restrict__ wqkv,
                                                    const float* __restrict__ wproj,
                                                    unsigned short* __restrict__ wsT) {
  int e = blockIdx.x * 256 + threadIdx.x;
  if (e < 288 * 96) {
    int n = e / 96, k = e - n * 96;
    wsT[e] = f2bf(wqkv[k * 288 + n]);
  } else {
    int e2 = e - 288 * 96;
    if (e2 < 96 * 96) {
      int n = e2 / 96, k = e2 - n * 96;
      wsT[288 * 96 + e2] = f2bf(wproj[k * 96 + n]);
    }
  }
}

__global__ __launch_bounds__(256) void swin_attn_kernel(const float* __restrict__ x,
                                                        const unsigned short* __restrict__ wT,
                                                        const float* __restrict__ bproj,
                                                        float* __restrict__ out) {
  __shared__ __align__(16) unsigned char smem[SMEM_BYTES];
  unsigned short* WQT = (unsigned short*)(smem + OFF_WQT);
  unsigned short* WPT = (unsigned short*)(smem + OFF_WPT);
  unsigned short* QKV = (unsigned short*)(smem + OFF_QKV);
  unsigned short* Xl  = (unsigned short*)(smem + OFF_R1);   // X (bf16), dead after QKV
  unsigned short* Pl  = (unsigned short*)(smem + OFF_R1);   // reused as P after barrier
  unsigned short* Ol  = (unsigned short*)(smem + OFF_O);

  const int tid  = threadIdx.x;
  const int lane = tid & 31;
  const int wv   = tid >> 5;       // 8 waves
  const int hf   = lane >> 4;
  const int wid  = blockIdx.x;

  // ---- async copy both transposed weight blocks (73728 B) into LDS ----
  {
    unsigned ldsb = (unsigned)(size_t)smem;
    const unsigned char* g = (const unsigned char*)wT;
#pragma unroll
    for (int it = 0; it < 18; ++it) {
      int ofs = (it * 256 + tid) * 16;
      unsigned la = ldsb + (unsigned)ofs;
      unsigned long long ga = (unsigned long long)(size_t)(g + ofs);
      asm volatile("global_load_async_to_lds_b128 %0, %1, off" :: "v"(la), "v"(ga) : "memory");
    }
  }
  // ---- stage X window as bf16 ----
  for (int e = tid; e < SLEN * C_; e += 256) {
    int t = e / C_, c = e - t * C_;
    Xl[t * XS + c] = f2bf(x[gidx(wid, t, c)]);
  }
  asm volatile("s_wait_asynccnt 0" ::: "memory");
  __syncthreads();

  // ---- QKV[64,288] = X[64,96] @ Wqkv : 72 tiles / 8 waves, K=96 in 3 steps ----
  // unroll 3 tiles per iteration: independent WMMA chains fill the
  // BF16 WMMA->WMMA RAW hazard slots (ISA 7.12.1) instead of v_nops.
#pragma unroll 3
  for (int i = 0; i < 9; ++i) {
    int tt = wv + 8 * i;
    int tm = tt & 3, tn = tt >> 2;
    v8f acc = {0.f, 0.f, 0.f, 0.f, 0.f, 0.f, 0.f, 0.f};
#pragma unroll
    for (int ks = 0; ks < 3; ++ks) {
      v16bf a = ldsA(Xl, XS, tm * 16, ks * 32, 32, lane);
      v16bf b = ldsBT(WQT, 96, tn * 16, ks * 32, 32, lane);
      acc = wmma_bf16(a, b, acc);
    }
    int col = tn * 16 + (lane & 15);
#pragma unroll
    for (int v = 0; v < 8; ++v)
      QKV[(tm * 16 + v + hf * 8) * QS + col] = f2bf(acc[v]);
  }
  __syncthreads();

  // ---- attention: 2 waves per head; this wave owns 32 rows (mh half) ----
  const int hd = wv >> 1;
  const int mh = wv & 1;
  unsigned short* Ph = Pl + hd * (SLEN * PS);
  {
    v8f S[2][4];
#pragma unroll
    for (int mi = 0; mi < 2; ++mi)
#pragma unroll
      for (int ni = 0; ni < 4; ++ni) {
        v8f acc = {0.f, 0.f, 0.f, 0.f, 0.f, 0.f, 0.f, 0.f};
        v16bf a = ldsA(QKV, QS, (mh * 2 + mi) * 16, hd * CH, CH, lane);          // Q, pad K->32
        v16bf b = ldsBT(QKV, QS, ni * 16, C_ + hd * CH, CH, lane);               // K^T
        S[mi][ni] = wmma_bf16(a, b, acc);
      }
    const float scl = 0.2041241452319315f;  // 1/sqrt(24)
#pragma unroll
    for (int mi = 0; mi < 2; ++mi) {
#pragma unroll
      for (int v = 0; v < 8; ++v) {
        float a0 = S[mi][0][v] * scl, a1 = S[mi][1][v] * scl;
        float a2 = S[mi][2][v] * scl, a3 = S[mi][3][v] * scl;
        float mx = fmaxf(fmaxf(a0, a1), fmaxf(a2, a3));
#pragma unroll
        for (int o = 1; o < 16; o <<= 1) mx = fmaxf(mx, __shfl_xor(mx, o, 32));
        float e0 = __expf(a0 - mx), e1 = __expf(a1 - mx);
        float e2 = __expf(a2 - mx), e3 = __expf(a3 - mx);
        float sm = (e0 + e1) + (e2 + e3);
#pragma unroll
        for (int o = 1; o < 16; o <<= 1) sm += __shfl_xor(sm, o, 32);
        float inv = 1.0f / sm;
        unsigned short* pr = Ph + (mh * 32 + mi * 16 + v + hf * 8) * PS + (lane & 15);
        pr[0]  = f2bf(e0 * inv);
        pr[16] = f2bf(e1 * inv);
        pr[32] = f2bf(e2 * inv);
        pr[48] = f2bf(e3 * inv);
      }
    }
  }
  __syncthreads();

  // ---- O[64,24] = P[64,64] @ V[64,24] (N padded to 32) ----
#pragma unroll
  for (int mi = 0; mi < 2; ++mi)
#pragma unroll
    for (int tnv = 0; tnv < 2; ++tnv) {
      v8f acc = {0.f, 0.f, 0.f, 0.f, 0.f, 0.f, 0.f, 0.f};
      int nrel = tnv * 16 + (lane & 15);
      int valid = nrel < CH;
#pragma unroll
      for (int ks = 0; ks < 2; ++ks) {
        v16bf a = ldsA(Ph, PS, mh * 32 + mi * 16, ks * 32, 32, lane);
        v16bf b = ldsBrm(QKV, QS, ks * 32, 2 * C_ + hd * CH + nrel, valid, lane);
        acc = wmma_bf16(a, b, acc);
      }
      if (valid) {
#pragma unroll
        for (int v = 0; v < 8; ++v)
          Ol[(mh * 32 + mi * 16 + v + hf * 8) * XS + hd * CH + nrel] = f2bf(acc[v]);
      }
    }
  __syncthreads();

  // ---- Y[64,96] = O @ Wproj + b ; 24 tiles / 8 waves ----
#pragma unroll
  for (int i = 0; i < 3; ++i) {
    int tt = wv + 8 * i;
    int tm = tt & 3, tn = tt >> 2;
    v8f acc = {0.f, 0.f, 0.f, 0.f, 0.f, 0.f, 0.f, 0.f};
#pragma unroll
    for (int ks = 0; ks < 3; ++ks) {
      v16bf a = ldsA(Ol, XS, tm * 16, ks * 32, 32, lane);
      v16bf b = ldsBT(WPT, 96, tn * 16, ks * 32, 32, lane);
      acc = wmma_bf16(a, b, acc);
    }
    int col = tn * 16 + (lane & 15);
    float bias = bproj[col];
#pragma unroll
    for (int v = 0; v < 8; ++v) {
      int t = tm * 16 + v + hf * 8;
      out[gidx(wid, t, col)] = acc[v] + bias;
    }
  }
}

extern "C" void kernel_launch(void* const* d_in, const int* in_sizes, int n_in,
                              void* d_out, int out_size, void* d_ws, size_t ws_size,
                              hipStream_t stream) {
  (void)in_sizes; (void)n_in; (void)out_size; (void)ws_size;
  const float* x      = (const float*)d_in[0];
  const float* w_qkv  = (const float*)d_in[1];
  const float* w_proj = (const float*)d_in[2];
  const float* b_proj = (const float*)d_in[3];
  unsigned short* wsT = (unsigned short*)d_ws;   // 73728 B used

  prep_weights<<<144, 256, 0, stream>>>(w_qkv, w_proj, wsT);
  swin_attn_kernel<<<NWIN, 256, 0, stream>>>(x, wsT, b_proj, (float*)d_out);
}
